// CrfRnnLayerSP_51582557224985
// MI455X (gfx1250) — matrix operand order repaired and checked by reading, not compiled, not run on hardware
//
#include <hip/hip_runtime.h>

// CRF-RNN mean-field layer for MI455X (gfx1250, wave32).
// Planar (C, H*W) fp32 intermediates kept L2-resident (~78 MB < 192 MB L2),
// bilateral weights precomputed once, message-passing 21x21 matmuls fused into
// two matrices (comp@Wsp, comp@Wbi) and executed with v_wmma_f32_16x16x32_f16.
// Each wave loads the constant A fragments once and sweeps 4 consecutive
// 16-pixel N-tiles (16 WMMAs/wave), with the q-update epilogue fused in-register
// via the documented 16x16 f32 C/D lane layout. All divides use v_rcp_f32.

typedef __attribute__((ext_vector_type(16))) _Float16 v16h;
typedef __attribute__((ext_vector_type(8)))  float    v8f;

static constexpr int   C  = 21;
static constexpr int   H  = 384;
static constexpr int   W  = 384;
static constexpr int   HW = H * W;
static constexpr int   NUM_ITER = 5;
static constexpr int   TPW = 4;            // N-tiles per wave in WMMA kernel
static constexpr float N_CLIQUES = 5.0f;
static constexpr float INV_2TA2 = 1.0f / (2.0f * 160.0f * 160.0f); // theta_alpha
static constexpr float INV_2TB2 = 1.0f / (2.0f * 3.0f * 3.0f);     // theta_beta

// Gaussian taps exp(-o^2 / (2*theta_gamma^2)), theta_gamma=3, o=-6..6
__constant__ float GK[13] = {
    0.13533528f, 0.24935221f, 0.41111229f, 0.60653067f, 0.80073740f,
    0.94595947f, 1.00000000f, 0.94595947f, 0.80073740f, 0.60653067f,
    0.41111229f, 0.24935221f, 0.13533528f};

__constant__ int CLIQ[5] = {5, 37, 81, 150, 230};

__device__ __forceinline__ float fastrcp(float x) {
    return __builtin_amdgcn_rcpf(x);       // v_rcp_f32
}

// ---------------------------------------------------------------- prep -----
// Asp = comp @ Wsp, Abi = comp @ Wbi, padded to 32x32 f16; norm1d for the
// separable spatial normalizer (blur of ones) = norm1d[y]*norm1d[x].
__global__ void prep_kernel(const float* __restrict__ Wsp,
                            const float* __restrict__ Wbi,
                            const float* __restrict__ Cm,
                            _Float16* __restrict__ Asp,
                            _Float16* __restrict__ Abi,
                            float* __restrict__ rnorm1d) {
    int idx = blockIdx.x * blockDim.x + threadIdx.x;
    if (idx < 32 * 32) {
        int r = idx >> 5, k = idx & 31;
        float as = 0.f, ab = 0.f;
        if (r < C && k < C) {
            for (int j = 0; j < C; ++j) {
                float cm = Cm[r * C + j];
                as += cm * Wsp[j * C + k];
                ab += cm * Wbi[j * C + k];
            }
        }
        Asp[idx] = (_Float16)as;
        Abi[idx] = (_Float16)ab;
    }
    if (idx < H) {
        float s = 0.f;
        #pragma unroll
        for (int i = 0; i < 13; ++i) {
            int yy = idx - (i - 6);
            if (yy >= 0 && yy < H) s += GK[i];
        }
        rnorm1d[idx] = fastrcp(s);          // store reciprocal
    }
}

// ------------------------------------------------- HWC -> planar, q = u ----
__global__ void init_kernel(const float* __restrict__ unaries,
                            float* __restrict__ u, float* __restrict__ q) {
    int idx = blockIdx.x * blockDim.x + threadIdx.x;   // idx over C*HW
    int c = idx / HW, p = idx - c * HW;
    float v = unaries[(size_t)p * C + c];
    u[idx] = v;
    q[idx] = v;
}

// -------------------------------------- bilateral weights (iter-invariant) -
__global__ void bilat_w_kernel(const float* __restrict__ rgb,
                               float* __restrict__ wbuf,
                               float* __restrict__ rbnorm) {
    int p = blockIdx.x * blockDim.x + threadIdx.x;
    if (p >= HW) return;
    int y = p / W, x = p - y * W;
    float r0 = rgb[(size_t)p * 3 + 0];
    float g0 = rgb[(size_t)p * 3 + 1];
    float b0 = rgb[(size_t)p * 3 + 2];
    float nsum = 0.f;
    #pragma unroll
    for (int t = 0; t < 25; ++t) {
        int dy = t / 5 - 2, dx = t % 5 - 2;
        int ny = y - dy, nx = x - dx;
        float wgt = 0.f;
        if (ny >= 0 && ny < H && nx >= 0 && nx < W) {
            int np = ny * W + nx;
            float dr = r0 - rgb[(size_t)np * 3 + 0];
            float dg = g0 - rgb[(size_t)np * 3 + 1];
            float db = b0 - rgb[(size_t)np * 3 + 2];
            float sw = __expf(-(float)(dy * dy + dx * dx) * INV_2TA2);
            float cw = __expf(-(dr * dr + dg * dg + db * db) * INV_2TB2);
            wgt = sw * cw;
        }
        wbuf[(size_t)t * HW + p] = wgt;
        nsum += wgt;
    }
    rbnorm[p] = fastrcp(nsum);              // store reciprocal
}

// ------------------------------------------------- msum (transposed map) ---
__global__ void msum_kernel(const int* __restrict__ spx,
                            float* __restrict__ msum) {
    int p = blockIdx.x * blockDim.x + threadIdx.x;
    if (p >= HW) return;
    int y = p / W, x = p - y * W;
    int sp = spx[x * W + y];            // sp_map = transpose(superpixels[0])
    float m = 0.f;
    #pragma unroll
    for (int i = 0; i < 5; ++i) m += (sp == CLIQ[i]) ? 1.f : 0.f;
    msum[p] = m;
}

// ---------------------------------------------------------- softmax (ch) ---
__global__ void softmax_kernel(const float* __restrict__ q,
                               float* __restrict__ sm) {
    int p = blockIdx.x * blockDim.x + threadIdx.x;
    if (p >= HW) return;
    float mx = -3.4e38f;
    #pragma unroll
    for (int c = 0; c < C; ++c) mx = fmaxf(mx, q[c * HW + p]);
    float s = 0.f;
    float e[C];
    #pragma unroll
    for (int c = 0; c < C; ++c) { e[c] = __expf(q[c * HW + p] - mx); s += e[c]; }
    float inv = fastrcp(s);
    #pragma unroll
    for (int c = 0; c < C; ++c) sm[c * HW + p] = e[c] * inv;
}

// -------------------------------------------------- spatial blur, y pass ---
__global__ void blur_y_kernel(const float* __restrict__ sm,
                              float* __restrict__ tmp) {
    int idx = blockIdx.x * blockDim.x + threadIdx.x;   // over C*HW
    int c = idx / HW, p = idx - c * HW;
    int y = p / W, x = p - y * W;
    float acc = 0.f;
    #pragma unroll
    for (int i = 0; i < 13; ++i) {
        int yy = y - (i - 6);
        if (yy >= 0 && yy < H) acc += GK[i] * sm[c * HW + yy * W + x];
    }
    tmp[idx] = acc;
}

// ------------------------------------- spatial blur, x pass + normalize ----
__global__ void blur_x_kernel(const float* __restrict__ tmp,
                              const float* __restrict__ rnorm1d,
                              float* __restrict__ sout) {
    int idx = blockIdx.x * blockDim.x + threadIdx.x;   // over C*HW
    int c = idx / HW, p = idx - c * HW;
    int y = p / W, x = p - y * W;
    float acc = 0.f;
    #pragma unroll
    for (int i = 0; i < 13; ++i) {
        int xx = x - (i - 6);
        if (xx >= 0 && xx < W) acc += GK[i] * tmp[c * HW + y * W + xx];
    }
    sout[idx] = acc * (rnorm1d[y] * rnorm1d[x]);
}

// --------------------------------------------- bilateral apply + normalize -
__global__ void bilat_apply_kernel(const float* __restrict__ sm,
                                   const float* __restrict__ wbuf,
                                   const float* __restrict__ rbnorm,
                                   float* __restrict__ bout) {
    int p = blockIdx.x * blockDim.x + threadIdx.x;
    if (p >= HW) return;
    int y = p / W, x = p - y * W;
    float acc[C];
    #pragma unroll
    for (int c = 0; c < C; ++c) acc[c] = 0.f;
    for (int t = 0; t < 25; ++t) {
        float wgt = wbuf[(size_t)t * HW + p];
        int dy = t / 5 - 2, dx = t % 5 - 2;
        int ny = y - dy, nx = x - dx;
        if (wgt != 0.f && ny >= 0 && ny < H && nx >= 0 && nx < W) {
            int np = ny * W + nx;
            #pragma unroll
            for (int c = 0; c < C; ++c) acc[c] += wgt * sm[c * HW + np];
        }
    }
    float inv = rbnorm[p];
    #pragma unroll
    for (int c = 0; c < C; ++c) bout[c * HW + p] = acc[c] * inv;
}

// --------------- WMMA message passing + compatibility + fused q update -----
// pairwise = Asp @ sout + Abi @ bout  (Asp = comp@Wsp, Abi = comp@Wbi, f16)
// q       = u - pairwise + sp_update(q_old)
// Each wave loads A once, then sweeps TPW 16-pixel N-tiles: 4*TPW WMMAs/wave.
__global__ __launch_bounds__(256) void mp_wmma_kernel(
    const _Float16* __restrict__ Asp, const _Float16* __restrict__ Abi,
    const float* __restrict__ sout, const float* __restrict__ bout,
    const float* __restrict__ u, float* __restrict__ q,
    const float* __restrict__ msum,
    const float* __restrict__ low, const float* __restrict__ high) {
    int wave = blockIdx.x * (blockDim.x >> 5) + (threadIdx.x >> 5);
    int lane = threadIdx.x & 31;
    int lp   = lane & 15;
    int hi   = lane >> 4;          // 0 or 1 (half-wave)

    // --- A fragments (16x32 f16), loaded once per wave:
    //     row = Mtile*16 + (lane%16), K = e + (e>=8 ? 8 : 0) + (lane>=16 ? 8 : 0)
    v16h a0s, a0b, a1s, a1b;
    #pragma unroll
    for (int e = 0; e < 16; ++e) {
        int k  = e + ((e >= 8) ? 8 : 0) + hi * 8;
        int r0 = lp;        // M-tile 0: rows 0..15
        int r1 = 16 + lp;   // M-tile 1: rows 16..31 (>=21 zero-padded)
        a0s[e] = Asp[r0 * 32 + k];
        a0b[e] = Abi[r0 * 32 + k];
        a1s[e] = Asp[r1 * 32 + k];
        a1b[e] = Abi[r1 * 32 + k];
    }

    float hwt = high[0];
    int tile0 = wave * TPW;

    #pragma unroll
    for (int t = 0; t < TPW; ++t) {
        int pix = (tile0 + t) * 16 + lp;   // this lane's pixel (N index)

        // B fragments (32x16 f16): N=lane%16, K = e + (e>=8 ? 8 : 0) + (lane>=16 ? 8 : 0)
        v16h bs, bb;
        #pragma unroll
        for (int e = 0; e < 16; ++e) {
            int k = e + ((e >= 8) ? 8 : 0) + hi * 8;
            float vs = (k < C) ? sout[k * HW + pix] : 0.f;
            float vb = (k < C) ? bout[k * HW + pix] : 0.f;
            bs[e] = (_Float16)vs;
            bb[e] = (_Float16)vb;
        }

        v8f acc0 = {}, acc1 = {};
        acc0 = __builtin_amdgcn_wmma_f32_16x16x32_f16(false, a0s, false, bs, (short)0, acc0, false, false);
        acc0 = __builtin_amdgcn_wmma_f32_16x16x32_f16(false, a0b, false, bb, (short)0, acc0, false, false);
        acc1 = __builtin_amdgcn_wmma_f32_16x16x32_f16(false, a1s, false, bs, (short)0, acc1, false, false);
        acc1 = __builtin_amdgcn_wmma_f32_16x16x32_f16(false, a1b, false, bb, (short)0, acc1, false, false);

        // fused epilogue: C/D layout -> lane<16: M=i; lane>=16: M=8+i
        // first_term = (ms*q + (5-ms))/q = ms + (5-ms)*rcp(q)
        float ms = msum[pix];
        #pragma unroll
        for (int i = 0; i < 8; ++i) {
            int c0 = i + hi * 8;       // channel from acc0
            {
                float qo = q[c0 * HW + pix];
                float ft = ms + (N_CLIQUES - ms) * fastrcp(qo);
                float sp = low[c0] * ft + hwt * (1.f - ft);
                q[c0 * HW + pix] = u[c0 * HW + pix] - acc0[i] + sp;
            }
            int c1 = 16 + c0;          // channel from acc1 (only c1 < C valid)
            if (c1 < C) {
                float qo = q[c1 * HW + pix];
                float ft = ms + (N_CLIQUES - ms) * fastrcp(qo);
                float sp = low[c1] * ft + hwt * (1.f - ft);
                q[c1 * HW + pix] = u[c1 * HW + pix] - acc1[i] + sp;
            }
        }
    }
}

// ------------------------------------------------------- planar -> HWC -----
__global__ void writeout_kernel(const float* __restrict__ q,
                                float* __restrict__ out) {
    int idx = blockIdx.x * blockDim.x + threadIdx.x;   // over HW*C, (p, c) order
    int p = idx / C, c = idx - p * C;
    out[idx] = q[c * HW + p];
}

// ===========================================================================
extern "C" void kernel_launch(void* const* d_in, const int* in_sizes, int n_in,
                              void* d_out, int out_size, void* d_ws, size_t ws_size,
                              hipStream_t stream) {
    const float* unaries = (const float*)d_in[0];
    const float* rgb     = (const float*)d_in[1];
    const int*   spx     = (const int*)  d_in[2];
    const float* Wsp     = (const float*)d_in[3];
    const float* Wbi     = (const float*)d_in[4];
    const float* Cm      = (const float*)d_in[5];
    const float* low     = (const float*)d_in[6];
    const float* high    = (const float*)d_in[7];
    float* out = (float*)d_out;

    // workspace layout (floats); bout aliases tmp (bilateral runs after blur_x)
    float* fws    = (float*)d_ws;
    float* u      = fws;
    float* q      = u     + (size_t)C * HW;
    float* sm     = q     + (size_t)C * HW;
    float* tmp    = sm    + (size_t)C * HW;  // y-blur scratch, then bilateral out
    float* sout   = tmp   + (size_t)C * HW;
    float* wbuf   = sout  + (size_t)C * HW;  // 25 planes of HW
    float* rbnorm = wbuf  + (size_t)25 * HW;
    float* msum   = rbnorm + HW;
    float* rn1d   = msum  + HW;              // H floats (reciprocal norm)
    _Float16* Asp = (_Float16*)(rn1d + H);   // 32x32 f16
    _Float16* Abi = Asp + 32 * 32;
    float* bout = tmp;                       // alias

    const int TB = 256;
    const int gPix  = HW / TB;                       // 576
    const int gElem = (C * HW) / TB;                 // 12096
    const int gOut  = (HW * C) / TB;                 // 12096
    const int gWmma = (HW / 16) / TPW / (TB / 32);   // 9216/4/8 = 288 blocks

    prep_kernel<<<1, 1024, 0, stream>>>(Wsp, Wbi, Cm, Asp, Abi, rn1d);
    init_kernel<<<gElem, TB, 0, stream>>>(unaries, u, q);
    bilat_w_kernel<<<gPix, TB, 0, stream>>>(rgb, wbuf, rbnorm);
    msum_kernel<<<gPix, TB, 0, stream>>>(spx, msum);

    for (int it = 0; it < NUM_ITER; ++it) {
        softmax_kernel<<<gPix, TB, 0, stream>>>(q, sm);
        blur_y_kernel<<<gElem, TB, 0, stream>>>(sm, tmp);
        blur_x_kernel<<<gElem, TB, 0, stream>>>(tmp, rn1d, sout);
        bilat_apply_kernel<<<gPix, TB, 0, stream>>>(sm, wbuf, rbnorm, bout);
        mp_wmma_kernel<<<gWmma, TB, 0, stream>>>(Asp, Abi, sout, bout,
                                                 u, q, msum, low, high);
    }
    writeout_kernel<<<gOut, TB, 0, stream>>>(q, out);
}
